// CausalMultiHeadSelfAttention_51462298140717
// MI455X (gfx1250) — matrix operand order repaired and checked by reading, not compiled, not run on hardware
//
#include <hip/hip_runtime.h>
#include <hip/hip_bf16.h>

// ---------------------------------------------------------------------------
// Causal MHA for MI455X (gfx1250, wave32).  All GEMM stages use
// v_wmma_f32_16x16x32_bf16 (f32 accumulation).  Flash-attention streaming
// softmax; S^T formulation keeps softmax reductions in-lane.
// Round-3 fix: the GEMM k-loops (trip count 32, compile-time) are FULLY
// unrolled.  No loop-carried fragment phis -> register allocator places each
// fragment's two b128 loads directly into the adjacent 8-VGPR WMMA operand
// tuple (zero v_mov rotation copies), and the scheduler pipelines loads
// across WMMA groups freely.
// ---------------------------------------------------------------------------

typedef __attribute__((ext_vector_type(16))) __bf16 v16bf;
typedef __attribute__((ext_vector_type(8)))  __bf16 v8bf;
typedef __attribute__((ext_vector_type(4)))  __bf16 v4bf;
typedef __attribute__((ext_vector_type(8)))  float  v8f;

#define D_MODEL 1024
#define S_LEN   2048
#define NHEAD   16
#define DK      64
#define BATCH   2
#define NTOK    (BATCH * S_LEN)   // 4096

static __device__ __forceinline__ v8f zero8() {
  v8f z;
#pragma unroll
  for (int i = 0; i < 8; ++i) z[i] = 0.0f;
  return z;
}

static __device__ __forceinline__ v8bf ldg8(const __bf16* p) {
  return *reinterpret_cast<const v8bf*>(p);   // 16B global_load_b128
}

static __device__ __forceinline__ v16bf combine16(v8bf lo, v8bf hi) {
  union { v16bf v; v8bf h[2]; } u;
  u.h[0] = lo; u.h[1] = hi;
  return u.v;
}

static __device__ __forceinline__ v8f wmma_bf16(v16bf a, v16bf b, v8f c) {
  return __builtin_amdgcn_wmma_f32_16x16x32_bf16(false, a, false, b,
                                                 (short)0, c, false, false);
}

// A-fragment: 16x32 bf16 tile of a row-major matrix, rows = lane15,
// k runs [k, k+8) and [k+16, k+24) per half-wave.
static __device__ __forceinline__ v16bf load_afrag(const __bf16* row, int kk,
                                                   int aoff) {
  return combine16(ldg8(row + kk + aoff), ldg8(row + kk + 16 + aoff));
}
// B-fragment: 32x16 bf16 tile of W^T (W row-major), k runs [k, k+8) and
// [k+16, k+24) per half-wave (same order as A).
static __device__ __forceinline__ v16bf load_bfrag(const __bf16* row, int kk) {
  return combine16(ldg8(row + kk), ldg8(row + kk + 16));
}

// ---------------------------------------------------------------------------
// f32 -> bf16 conversion (4 elements/thread, all sizes are multiples of 4)
// ---------------------------------------------------------------------------
__global__ void cvt_bf16_kernel(const float* __restrict__ in,
                                __bf16* __restrict__ out, int n) {
  int i = (blockIdx.x * blockDim.x + threadIdx.x) * 4;
  if (i < n) {
    float4 f = *reinterpret_cast<const float4*>(in + i);
    v4bf o;
    o[0] = (__bf16)f.x; o[1] = (__bf16)f.y;
    o[2] = (__bf16)f.z; o[3] = (__bf16)f.w;
    *reinterpret_cast<v4bf*>(out + i) = o;
  }
}

// ---------------------------------------------------------------------------
// Stage 1: fused QKV projection + RoPE.
//   grid = (NTOK/16, 3072/512), block = 256 (8 waves).
//   Wave w handles a 16x64 output tile of the fused [Q|K|V] (N = 3072) GEMM;
//   the K=1024 loop is fully unrolled (32 WMMA steps, 4 tiles each).
//   Q/K: RoPE applied, bf16 store to [B,H,S,DK] (Q pre-scaled by
//        0.125*log2(e) so attention softmax can use exp2).
//   V:   stored transposed [B,H,DK,S] -> contiguous b128 stores, and makes
//        the PV-matmul A-fragments contiguous b128 loads.
// ---------------------------------------------------------------------------
__global__ __launch_bounds__(256) void qkv_rope_kernel(
    const __bf16* __restrict__ xb,
    const __bf16* __restrict__ wqb,
    const __bf16* __restrict__ wkb,
    const __bf16* __restrict__ wvb,
    const int*    __restrict__ tpos,
    __bf16* __restrict__ Q,    // [B,H,S,DK]
    __bf16* __restrict__ K,    // [B,H,S,DK]
    __bf16* __restrict__ Vt)   // [B,H,DK,S]
{
  const int lane   = threadIdx.x & 31;
  const int wave   = threadIdx.x >> 5;
  const int lane15 = lane & 15;
  const int hi     = lane >> 4;

  const int m0    = blockIdx.x * 16;                 // token tile
  const int ncol0 = blockIdx.y * 512 + wave * 64;    // fused col in [0,3072)
  const int mat   = ncol0 >> 10;                     // 0=Q 1=K 2=V (uniform)
  const int nloc0 = ncol0 & 1023;

  const __bf16* wb = (mat == 0) ? wqb : (mat == 1) ? wkb : wvb;

  v8f acc[4];
#pragma unroll
  for (int t = 0; t < 4; ++t) acc[t] = zero8();

  const __bf16* arow = xb + (size_t)(m0 + lane15) * D_MODEL;
  const int aoff = hi ? 8 : 0;
  const int boff = hi ? 8 : 0;
  const __bf16* brow[4];
#pragma unroll
  for (int t = 0; t < 4; ++t)
    brow[t] = wb + (size_t)(nloc0 + t * 16 + lane15) * D_MODEL + boff;

  // fully unrolled K loop: no loop-carried fragment registers
#pragma unroll
  for (int kk = 0; kk < D_MODEL; kk += 32) {
    v16bf afrag = load_afrag(arow, kk, aoff);
#pragma unroll
    for (int t = 0; t < 4; ++t) {
      v16bf bfrag = load_bfrag(brow[t], kk);
      acc[t] = wmma_bf16(afrag, bfrag, acc[t]);
    }
  }

  const int tok0  = m0 + (hi ? 8 : 0);   // C-layout: lane holds 8 rows
  const int b     = tok0 >> 11;          // / S_LEN
  const int srow0 = tok0 & (S_LEN - 1);

  if (mat < 2) {
    // RoPE + store Q/K.  Pair partner lives in the adjacent lane.
    const float qscale = 0.18033688011112042f;  // 0.125 * log2(e)
    __bf16* dst = (mat == 0) ? Q : K;
#pragma unroll
    for (int t = 0; t < 4; ++t) {
      const int col = nloc0 + t * 16 + lane15;
      const int h = col >> 6, d = col & 63;
      // inv_freq = theta^(-(2i)/D), 2i = col & ~1 ; log2(10000)=13.2877124
      const float inv_freq =
          exp2f(-(float)(col & ~1) * (13.287712379549449f / 1024.0f));
#pragma unroll
      for (int r = 0; r < 8; ++r) {
        const float pos = (float)tpos[tok0 + r];
        float sn, cs;
        __sincosf(pos * inv_freq, &sn, &cs);
        const float mine  = acc[t][r];
        const float other = __shfl_xor(mine, 1, 32);
        float v = (col & 1) ? (other * sn + mine * cs)   // r2 = x1 s + x2 c
                            : (mine * cs - other * sn);  // r1 = x1 c - x2 s
        if (mat == 0) v *= qscale;
        dst[((size_t)(b * NHEAD + h) * S_LEN + (srow0 + r)) * DK + d] =
            (__bf16)v;
      }
    }
  } else {
    // V transposed store: lane's 8 rows are 8 consecutive tokens at fixed d
    // -> one b128 store per tile.
#pragma unroll
    for (int t = 0; t < 4; ++t) {
      const int col = nloc0 + t * 16 + lane15;
      const int h = col >> 6, d = col & 63;
      v8bf ov;
#pragma unroll
      for (int r = 0; r < 8; ++r) ov[r] = (__bf16)acc[t][r];
      *reinterpret_cast<v8bf*>(
          Vt + ((size_t)(b * NHEAD + h) * DK + d) * S_LEN + srow0) = ov;
    }
  }
}

// ---------------------------------------------------------------------------
// Stage 2: causal flash attention, one wave per (b,h,16-query tile).
//   Computes S^T = K * Q^T (so the kv softmax axis is in-lane in C-layout),
//   online softmax in base-2 (scale folded into Q), then O^T = V^T * P^T.
//   P^T C-layout -> B-operand needs only a half-wave swap (shfl_xor 16).
//   All K and V^T fragments of the current kv block are loaded up front so
//   the V loads overlap the score WMMAs and the softmax VALU work; block-
//   level latency is otherwise hidden by wave-level parallelism (4096
//   single-wave workgroups).
// ---------------------------------------------------------------------------
__global__ __launch_bounds__(32) void attn_kernel(
    const __bf16* __restrict__ Q,   // [B,H,S,DK], pre-scaled
    const __bf16* __restrict__ K,   // [B,H,S,DK]
    const __bf16* __restrict__ Vt,  // [B,H,DK,S]
    __bf16* __restrict__ O)         // [B,S,D]
{
  const int lane   = threadIdx.x & 31;
  const int lane15 = lane & 15;
  const int hi     = lane >> 4;
  const int qt = blockIdx.x;
  const int bh = blockIdx.y;
  const int q0 = qt * 16;

  const __bf16* Qb = Q  + (size_t)bh * S_LEN * DK;
  const __bf16* Kb = K  + (size_t)bh * S_LEN * DK;
  const __bf16* Vb = Vt + (size_t)bh * DK * S_LEN;

  // Q^T B-fragments (dk split into two K=32 steps), loaded once.
  v16bf qfrag[2];
#pragma unroll
  for (int ks = 0; ks < 2; ++ks) {
    const __bf16* p = Qb + (size_t)(q0 + lane15) * DK + ks * 32 + (hi ? 8 : 0);
    qfrag[ks] = combine16(ldg8(p), ldg8(p + 16));
  }

  float m_run = -1e30f, l_run = 0.0f;
  v8f oacc[4];
#pragma unroll
  for (int c = 0; c < 4; ++c) oacc[c] = zero8();

  const int nblk = ((q0 + 15) >> 5) + 1;   // kv blocks of 32, causal bound
  const int qcol = q0 + lane15;            // this lane's query column
  const int roff = hi ? 8 : 0;

  for (int ib = 0; ib < nblk; ++ib) {
    const int kv0 = ib * 32;

    // --- issue ALL independent loads for this block first ------------------
    // V^T A-fragments (4 chunks of 16 d-rows) -- consumed only after softmax,
    // so these 8 b128 loads hide behind the score WMMAs + exp2 work.
    v16bf vf[4];
#pragma unroll
    for (int c = 0; c < 4; ++c) {
      const __bf16* vp = Vb + (size_t)(c * 16 + lane15) * S_LEN + kv0 + (hi ? 16 : 0);
      vf[c] = combine16(ldg8(vp), ldg8(vp + 8));
    }
    // K A-fragments: [tile 0/1][dk step 0/1]
    v16bf kf[2][2];
#pragma unroll
    for (int ks = 0; ks < 2; ++ks) {
      const __bf16* k0p = Kb + (size_t)(kv0 + lane15) * DK      + ks * 32 + roff;
      const __bf16* k1p = Kb + (size_t)(kv0 + 16 + lane15) * DK + ks * 32 + roff;
      kf[0][ks] = combine16(ldg8(k0p), ldg8(k0p + 16));
      kf[1][ks] = combine16(ldg8(k1p), ldg8(k1p + 16));
    }

    // --- scores: S^T tiles (kv x q), 16 kv rows each ---
    v8f c0 = zero8(), c1 = zero8();
#pragma unroll
    for (int ks = 0; ks < 2; ++ks) {
      c0 = wmma_bf16(kf[0][ks], qfrag[ks], c0);
      c1 = wmma_bf16(kf[1][ks], qfrag[ks], c1);
    }

    // --- causal mask + column max (kv axis is in-lane + partner lane) ---
    float mloc = -1e30f;
#pragma unroll
    for (int r = 0; r < 8; ++r) {
      if (kv0 + roff + r      > qcol) c0[r] = -1e30f;
      if (kv0 + 16 + roff + r > qcol) c1[r] = -1e30f;
      mloc = fmaxf(mloc, fmaxf(c0[r], c1[r]));
    }
    const float mnew  = fmaxf(m_run, fmaxf(mloc, __shfl_xor(mloc, 16, 32)));
    const float alpha = exp2f(m_run - mnew);   // base-2: scale folded into Q
    m_run = mnew;

    float p0[8], p1[8], ssum = 0.0f;
#pragma unroll
    for (int r = 0; r < 8; ++r) {
      p0[r] = exp2f(c0[r] - mnew);
      p1[r] = exp2f(c1[r] - mnew);
      ssum += p0[r] + p1[r];
    }
    l_run = alpha * l_run + ssum + __shfl_xor(ssum, 16, 32);

    // rescale running output (alpha uniform per lane: q = column = lane)
#pragma unroll
    for (int c = 0; c < 4; ++c)
#pragma unroll
      for (int r = 0; r < 8; ++r) oacc[c][r] *= alpha;

    // --- P^T (C-layout) -> B-operand (K=32 over kv): half-wave swap ---
    v16bf pb;
#pragma unroll
    for (int r = 0; r < 8; ++r) {
      const float x0 = __shfl_xor(p0[r], 16, 32);
      const float x1 = __shfl_xor(p1[r], 16, 32);
      pb[r]     = (__bf16)(hi ? x1    : p0[r]);
      pb[r + 8] = (__bf16)(hi ? p1[r] : x0);
    }

    // --- O^T += V^T * P^T (4 chunks of 16 d-rows) ---
#pragma unroll
    for (int c = 0; c < 4; ++c) oacc[c] = wmma_bf16(vf[c], pb, oacc[c]);
  }

  // normalize and store O (row-major [B,S,D]); 8 consecutive d per lane.
  const float invl = 1.0f / l_run;
  const int b = bh >> 4, h = bh & 15;
  const int tok = q0 + lane15;
#pragma unroll
  for (int c = 0; c < 4; ++c) {
    v8bf ov;
#pragma unroll
    for (int r = 0; r < 8; ++r) ov[r] = (__bf16)(oacc[c][r] * invl);
    *reinterpret_cast<v8bf*>(
        O + ((size_t)b * S_LEN + tok) * D_MODEL + h * DK + c * 16 + roff) = ov;
  }
}

// ---------------------------------------------------------------------------
// Stage 3: output projection  out = O @ wo^T  (f32 output), fully unrolled
// K loop like stage 1.
// ---------------------------------------------------------------------------
__global__ __launch_bounds__(256) void oproj_kernel(
    const __bf16* __restrict__ Ob,    // [NTOK, D]
    const __bf16* __restrict__ wob,   // [D, D]
    float* __restrict__ out)          // [NTOK, D] f32
{
  const int lane   = threadIdx.x & 31;
  const int wave   = threadIdx.x >> 5;
  const int lane15 = lane & 15;
  const int hi     = lane >> 4;

  const int m0 = blockIdx.x * 16;
  const int n0 = blockIdx.y * 512 + wave * 64;

  v8f acc[4];
#pragma unroll
  for (int t = 0; t < 4; ++t) acc[t] = zero8();

  const __bf16* arow = Ob + (size_t)(m0 + lane15) * D_MODEL;
  const int aoff = hi ? 8 : 0;
  const int boff = hi ? 8 : 0;
  const __bf16* brow[4];
#pragma unroll
  for (int t = 0; t < 4; ++t)
    brow[t] = wob + (size_t)(n0 + t * 16 + lane15) * D_MODEL + boff;

#pragma unroll
  for (int kk = 0; kk < D_MODEL; kk += 32) {
    v16bf afrag = load_afrag(arow, kk, aoff);
#pragma unroll
    for (int t = 0; t < 4; ++t) {
      v16bf bfrag = load_bfrag(brow[t], kk);
      acc[t] = wmma_bf16(afrag, bfrag, acc[t]);
    }
  }

  const int tok0 = m0 + (hi ? 8 : 0);
#pragma unroll
  for (int t = 0; t < 4; ++t) {
    const int col = n0 + t * 16 + lane15;
#pragma unroll
    for (int r = 0; r < 8; ++r)
      out[(size_t)(tok0 + r) * D_MODEL + col] = acc[t][r];
  }
}

// ---------------------------------------------------------------------------
extern "C" void kernel_launch(void* const* d_in, const int* in_sizes, int n_in,
                              void* d_out, int out_size, void* d_ws,
                              size_t ws_size, hipStream_t stream) {
  const float* x    = (const float*)d_in[0];
  const int*   tpos = (const int*)d_in[1];
  const float* wq   = (const float*)d_in[2];
  const float* wk   = (const float*)d_in[3];
  const float* wv   = (const float*)d_in[4];
  const float* wo   = (const float*)d_in[5];
  float* out = (float*)d_out;

  // Workspace carve-up (~48 MB total, 256B aligned slices).
  char* ws = (char*)d_ws;
  size_t off = 0;
  auto carve = [&](size_t bytes) -> void* {
    void* p = ws + off;
    off += (bytes + 255) & ~(size_t)255;
    return p;
  };
  __bf16* xb  = (__bf16*)carve((size_t)NTOK * D_MODEL * 2);
  __bf16* wqb = (__bf16*)carve((size_t)D_MODEL * D_MODEL * 2);
  __bf16* wkb = (__bf16*)carve((size_t)D_MODEL * D_MODEL * 2);
  __bf16* wvb = (__bf16*)carve((size_t)D_MODEL * D_MODEL * 2);
  __bf16* wob = (__bf16*)carve((size_t)D_MODEL * D_MODEL * 2);
  __bf16* Qs  = (__bf16*)carve((size_t)NTOK * D_MODEL * 2);
  __bf16* Ks  = (__bf16*)carve((size_t)NTOK * D_MODEL * 2);
  __bf16* Vt  = (__bf16*)carve((size_t)NTOK * D_MODEL * 2);
  __bf16* Oa  = (__bf16*)carve((size_t)NTOK * D_MODEL * 2);

  const int nx = NTOK * D_MODEL;        // 4M
  const int nw = D_MODEL * D_MODEL;     // 1M
  cvt_bf16_kernel<<<(nx / 4 + 255) / 256, 256, 0, stream>>>(x,  xb,  nx);
  cvt_bf16_kernel<<<(nw / 4 + 255) / 256, 256, 0, stream>>>(wq, wqb, nw);
  cvt_bf16_kernel<<<(nw / 4 + 255) / 256, 256, 0, stream>>>(wk, wkb, nw);
  cvt_bf16_kernel<<<(nw / 4 + 255) / 256, 256, 0, stream>>>(wv, wvb, nw);
  cvt_bf16_kernel<<<(nw / 4 + 255) / 256, 256, 0, stream>>>(wo, wob, nw);

  qkv_rope_kernel<<<dim3(NTOK / 16, 6), 256, 0, stream>>>(
      xb, wqb, wkb, wvb, tpos, Qs, Ks, Vt);

  attn_kernel<<<dim3(S_LEN / 16, BATCH * NHEAD), 32, 0, stream>>>(
      Qs, Ks, Vt, Oa);

  oproj_kernel<<<dim3(NTOK / 16, 2), 256, 0, stream>>>(Oa, wob, out);
}
